// LlamaAttentionNew_369367188109
// MI455X (gfx1250) — compile-verified, matrix operand order checked by reading, not run on hardware
//
#include <hip/hip_runtime.h>

#define S_LEN 2048
#define HDIM  4096
#define NH    32
#define NKV   8
#define DH    128
#define KVW   (NKV*DH)   // 1024

typedef __attribute__((ext_vector_type(16))) __bf16       v16bf;
typedef __attribute__((ext_vector_type(8)))  float        v8f;
typedef __attribute__((ext_vector_type(4)))  unsigned int v4u;

__device__ __forceinline__ unsigned short f2bf(float f) {
    unsigned int u = __float_as_uint(f);
    u += 0x7FFFu + ((u >> 16) & 1u);        // round-to-nearest-even
    return (unsigned short)(u >> 16);
}
__device__ __forceinline__ float bf2f(unsigned short h) {
    return __uint_as_float(((unsigned int)h) << 16);
}

// 16x32 bf16 fragment (A-layout / B^T-layout): lane L holds row (L&15),
// elements 0-7 = K [koff..koff+7], elements 8-15 = K [16+koff..16+koff+7],
// koff = (L>>4)*8.  Two contiguous 16B loads per lane.
union FragBF {
    v16bf v;
    v4u   q[2];
};

__device__ __forceinline__ void load_frag(FragBF& f, const unsigned short* p) {
    f.q[0] = *reinterpret_cast<const v4u*>(p);
    f.q[1] = *reinterpret_cast<const v4u*>(p + 16);
}

// ---------------------------------------------------------------------------
// f32 -> bf16 conversion
// ---------------------------------------------------------------------------
__global__ void cvt_bf16_kernel(const float* __restrict__ src,
                                unsigned short* __restrict__ dst, int n) {
    int i = blockIdx.x * blockDim.x + threadIdx.x;
    if (i < n) dst[i] = f2bf(src[i]);
}

// ---------------------------------------------------------------------------
// C[M,N] = A[M,K] * B[N,K]^T   (bf16 in, f32 accumulate, bf16 or f32 out)
// block = 256 threads = 8 waves (2 x 4); block tile 128x256; wave tile 64x64
// 16 WMMA per k-step vs 16 b128 loads -> 1:1 vmem:wmma issue ratio.
// ---------------------------------------------------------------------------
__global__ __launch_bounds__(256) void gemm_bf16_kernel(
    const unsigned short* __restrict__ A,
    const unsigned short* __restrict__ B,
    void* __restrict__ C, int M, int N, int K, int c_f32)
{
    const int lane = threadIdx.x & 31;
    const int wave = threadIdx.x >> 5;
    const int r    = lane & 15;
    const int hi   = lane >> 4;
    const int koff = hi * 8;
    const int wm   = wave >> 2;   // 0..1
    const int wn   = wave & 3;    // 0..3
    const int rowA = blockIdx.y * 128 + wm * 64;
    const int colB = blockIdx.x * 256 + wn * 64;

    v8f acc[4][4];
#pragma unroll
    for (int i = 0; i < 4; ++i)
#pragma unroll
        for (int j = 0; j < 4; ++j)
#pragma unroll
            for (int t = 0; t < 8; ++t) acc[i][j][t] = 0.0f;

    const unsigned short* Abase = A + (size_t)(rowA + r) * K + koff;
    const unsigned short* Bbase = B + (size_t)(colB + r) * K + koff;

    for (int k = 0; k < K; k += 32) {
        // Speculative prefetch ~16 k-steps ahead: streams weights HBM->GL2
        // on first pass (global_prefetch_b8), harmless L2 hit afterwards.
        if (k + 512 < K) {
            __builtin_prefetch(Abase + k + 512, 0, 1);
            __builtin_prefetch(Bbase + k + 512, 0, 1);
        }
        FragBF a[4];
#pragma unroll
        for (int i = 0; i < 4; ++i)
            load_frag(a[i], Abase + (size_t)i * 16 * K + k);
        FragBF b[4];
#pragma unroll
        for (int j = 0; j < 4; ++j)
            load_frag(b[j], Bbase + (size_t)j * 16 * K + k);
#pragma unroll
        for (int i = 0; i < 4; ++i)
#pragma unroll
            for (int j = 0; j < 4; ++j)
                acc[i][j] = __builtin_amdgcn_wmma_f32_16x16x32_bf16(
                    false, a[i].v, false, b[j].v, (short)0, acc[i][j],
                    false, false);
    }

    // C/D layout: VGPR t -> m = t + 8*hi ; n = lane&15
    if (c_f32) {
        float* Cf = (float*)C;
#pragma unroll
        for (int i = 0; i < 4; ++i)
#pragma unroll
            for (int j = 0; j < 4; ++j)
#pragma unroll
                for (int t = 0; t < 8; ++t) {
                    int row = rowA + i * 16 + t + 8 * hi;
                    int col = colB + j * 16 + r;
                    Cf[(size_t)row * N + col] = acc[i][j][t];
                }
    } else {
        unsigned short* Ch = (unsigned short*)C;
#pragma unroll
        for (int i = 0; i < 4; ++i)
#pragma unroll
            for (int j = 0; j < 4; ++j)
#pragma unroll
                for (int t = 0; t < 8; ++t) {
                    int row = rowA + i * 16 + t + 8 * hi;
                    int col = colB + j * 16 + r;
                    Ch[(size_t)row * N + col] = f2bf(acc[i][j][t]);
                }
    }
}

// ---------------------------------------------------------------------------
// RoPE (in-place on bf16 Q or K); rotate_half with half = 64
// ---------------------------------------------------------------------------
__global__ void rope_kernel(unsigned short* __restrict__ X,
                            const float* __restrict__ cs,
                            const float* __restrict__ sn,
                            int nh, int total)
{
    int idx = blockIdx.x * blockDim.x + threadIdx.x;
    if (idx >= total) return;
    int d = idx & 63;
    int h = (idx >> 6) % nh;
    int s = idx / (64 * nh);
    size_t base = (size_t)s * (nh * 128) + h * 128;
    float x0 = bf2f(X[base + d]);
    float x1 = bf2f(X[base + d + 64]);
    float c0 = cs[s * 128 + d],  c1 = cs[s * 128 + d + 64];
    float s0 = sn[s * 128 + d],  s1 = sn[s * 128 + d + 64];
    X[base + d]      = f2bf(x0 * c0 - x1 * s0);
    X[base + d + 64] = f2bf(x1 * c1 + x0 * s1);
}

// ---------------------------------------------------------------------------
// V[s, kvh*128+d] -> Vt[(kvh*128+d), s]  (so P*V B-fragments are contiguous)
// ---------------------------------------------------------------------------
__global__ void vtrans_kernel(const unsigned short* __restrict__ V,
                              unsigned short* __restrict__ Vt)
{
    int i = blockIdx.x * blockDim.x + threadIdx.x;
    if (i >= S_LEN * KVW) return;
    int c = i & (KVW - 1);
    int s = i >> 10;
    Vt[(size_t)c * S_LEN + s] = V[i];
}

// ---------------------------------------------------------------------------
// Flash attention, causal, GQA 4:1.
// grid = (S/64, NH); block = 128 = 4 waves; wave handles 16 q-rows.
// ---------------------------------------------------------------------------
__global__ __launch_bounds__(128) void attn_kernel(
    const unsigned short* __restrict__ Q,
    const unsigned short* __restrict__ Kb,
    const unsigned short* __restrict__ Vt,
    unsigned short* __restrict__ AO)
{
    __shared__ __align__(16) unsigned short pbuf[4][16 * 32];

    const int h    = blockIdx.y;
    const int kvh  = h >> 2;
    const int wave = threadIdx.x >> 5;
    const int lane = threadIdx.x & 31;
    const int r    = lane & 15;
    const int hi   = lane >> 4;
    const int koff = hi * 8;
    const int qbase = blockIdx.x * 64 + wave * 16;
    unsigned short* pb = &pbuf[wave][0];

    // Preload Q fragments for this wave's 16 rows (D = 4 x 32)
    FragBF qf[4];
#pragma unroll
    for (int dk = 0; dk < 4; ++dk)
        load_frag(qf[dk],
                  Q + (size_t)(qbase + r) * HDIM + h * DH + dk * 32 + koff);

    v8f o[8];
#pragma unroll
    for (int dt = 0; dt < 8; ++dt)
#pragma unroll
        for (int t = 0; t < 8; ++t) o[dt][t] = 0.0f;

    float Mrow[8], Lrow[8];
#pragma unroll
    for (int t = 0; t < 8; ++t) { Mrow[t] = -3.0e38f; Lrow[t] = 0.0f; }

    const float scale = 0.088388347648318447f;   // 1/sqrt(128)
    const int kend = qbase + 16;                 // causal extent (wave-uniform)

    for (int kt = 0; kt < kend; kt += 32) {
        // ---- S = Q * K^T (two 16-col n-tiles) ----
        v8f s0, s1;
#pragma unroll
        for (int t = 0; t < 8; ++t) { s0[t] = 0.0f; s1[t] = 0.0f; }
#pragma unroll
        for (int dk = 0; dk < 4; ++dk) {
            FragBF kf0, kf1;
            load_frag(kf0, Kb + (size_t)(kt      + r) * KVW + kvh * DH + dk * 32 + koff);
            load_frag(kf1, Kb + (size_t)(kt + 16 + r) * KVW + kvh * DH + dk * 32 + koff);
            s0 = __builtin_amdgcn_wmma_f32_16x16x32_bf16(
                false, qf[dk].v, false, kf0.v, (short)0, s0, false, false);
            s1 = __builtin_amdgcn_wmma_f32_16x16x32_bf16(
                false, qf[dk].v, false, kf1.v, (short)0, s1, false, false);
        }

        // ---- online softmax (row reductions via xor-shuffles inside halves) ----
        float alpha[8];
#pragma unroll
        for (int t = 0; t < 8; ++t) {
            int qi = qbase + t + 8 * hi;
            int k0 = kt + r, k1 = kt + 16 + r;
            float e0 = (k0 <= qi) ? s0[t] * scale : -3.0e38f;
            float e1 = (k1 <= qi) ? s1[t] * scale : -3.0e38f;
            float v = fmaxf(e0, e1);
            v = fmaxf(v, __shfl_xor(v, 1, 32));
            v = fmaxf(v, __shfl_xor(v, 2, 32));
            v = fmaxf(v, __shfl_xor(v, 4, 32));
            v = fmaxf(v, __shfl_xor(v, 8, 32));
            float newM = fmaxf(Mrow[t], v);
            alpha[t] = __expf(Mrow[t] - newM);
            Mrow[t] = newM;
            float p0 = __expf(e0 - newM);
            float p1 = __expf(e1 - newM);
            float ps = p0 + p1;
            ps += __shfl_xor(ps, 1, 32);
            ps += __shfl_xor(ps, 2, 32);
            ps += __shfl_xor(ps, 4, 32);
            ps += __shfl_xor(ps, 8, 32);
            Lrow[t] = Lrow[t] * alpha[t] + ps;
            int m = t + 8 * hi;
            pb[m * 32 + r]      = f2bf(p0);   // stage P for transpose
            pb[m * 32 + 16 + r] = f2bf(p1);
        }
#pragma unroll
        for (int dt = 0; dt < 8; ++dt)
#pragma unroll
            for (int t = 0; t < 8; ++t) o[dt][t] *= alpha[t];

        // ---- P (16x32 bf16, A-fragment layout from LDS) ----
        FragBF pf;
        load_frag(pf, pb + r * 32 + koff);

        // ---- O += P * V ----
#pragma unroll
        for (int dt = 0; dt < 8; ++dt) {
            FragBF vf;
            load_frag(vf, Vt + (size_t)(kvh * DH + dt * 16 + r) * S_LEN + kt + koff);
            o[dt] = __builtin_amdgcn_wmma_f32_16x16x32_bf16(
                false, pf.v, false, vf.v, (short)0, o[dt], false, false);
        }
    }

    float inv[8];
#pragma unroll
    for (int t = 0; t < 8; ++t) inv[t] = 1.0f / Lrow[t];
#pragma unroll
    for (int dt = 0; dt < 8; ++dt)
#pragma unroll
        for (int t = 0; t < 8; ++t)
            AO[(size_t)(qbase + t + 8 * hi) * HDIM + h * DH + dt * 16 + r] =
                f2bf(o[dt][t] * inv[t]);
}

// ---------------------------------------------------------------------------
extern "C" void kernel_launch(void* const* d_in, const int* in_sizes, int n_in,
                              void* d_out, int out_size, void* d_ws, size_t ws_size,
                              hipStream_t stream)
{
    (void)in_sizes; (void)n_in; (void)out_size; (void)ws_size;

    const float* hs = (const float*)d_in[0];
    const float* cs = (const float*)d_in[1];
    const float* sn = (const float*)d_in[2];
    const float* Wq = (const float*)d_in[3];
    const float* Wk = (const float*)d_in[4];
    const float* Wv = (const float*)d_in[5];
    const float* Wo = (const float*)d_in[6];

    char* ws = (char*)d_ws;
    size_t off = 0;
    auto alloc = [&](size_t bytes) -> void* {
        void* p = ws + off;
        off += (bytes + 255) & ~(size_t)255;
        return p;
    };
    unsigned short* hB  = (unsigned short*)alloc((size_t)S_LEN * HDIM * 2);
    unsigned short* WqB = (unsigned short*)alloc((size_t)HDIM * HDIM * 2);
    unsigned short* WkB = (unsigned short*)alloc((size_t)KVW * HDIM * 2);
    unsigned short* WvB = (unsigned short*)alloc((size_t)KVW * HDIM * 2);
    unsigned short* WoB = (unsigned short*)alloc((size_t)HDIM * HDIM * 2);
    unsigned short* Qw  = (unsigned short*)alloc((size_t)S_LEN * HDIM * 2);
    unsigned short* Kw  = (unsigned short*)alloc((size_t)S_LEN * KVW * 2);
    unsigned short* Vw  = (unsigned short*)alloc((size_t)S_LEN * KVW * 2);
    unsigned short* VtW = (unsigned short*)alloc((size_t)S_LEN * KVW * 2);
    unsigned short* AOw = (unsigned short*)alloc((size_t)S_LEN * HDIM * 2);

    auto cvt = [&](const float* s, unsigned short* d, int n) {
        cvt_bf16_kernel<<<(n + 255) / 256, 256, 0, stream>>>(s, d, n);
    };
    cvt(hs, hB,  S_LEN * HDIM);
    cvt(Wq, WqB, HDIM * HDIM);
    cvt(Wk, WkB, KVW * HDIM);
    cvt(Wv, WvB, KVW * HDIM);
    cvt(Wo, WoB, HDIM * HDIM);

    dim3 blk(256);
    gemm_bf16_kernel<<<dim3(HDIM / 256, S_LEN / 128), blk, 0, stream>>>(
        hB, WqB, Qw, S_LEN, HDIM, HDIM, 0);
    gemm_bf16_kernel<<<dim3(KVW / 256, S_LEN / 128), blk, 0, stream>>>(
        hB, WkB, Kw, S_LEN, KVW, HDIM, 0);
    gemm_bf16_kernel<<<dim3(KVW / 256, S_LEN / 128), blk, 0, stream>>>(
        hB, WvB, Vw, S_LEN, KVW, HDIM, 0);

    rope_kernel<<<(S_LEN * NH  * 64 + 255) / 256, 256, 0, stream>>>(
        Qw, cs, sn, NH, S_LEN * NH * 64);
    rope_kernel<<<(S_LEN * NKV * 64 + 255) / 256, 256, 0, stream>>>(
        Kw, cs, sn, NKV, S_LEN * NKV * 64);

    vtrans_kernel<<<(S_LEN * KVW + 255) / 256, 256, 0, stream>>>(Vw, VtW);

    attn_kernel<<<dim3(S_LEN / 64, NH), dim3(128), 0, stream>>>(Qw, Kw, VtW, AOw);

    gemm_bf16_kernel<<<dim3(HDIM / 256, S_LEN / 128), blk, 0, stream>>>(
        AOw, WoB, (float*)d_out, S_LEN, HDIM, HDIM, 1);
}
